// Mamba2EncoderLayer_22806276342239
// MI455X (gfx1250) — matrix operand order repaired and checked by reading, not compiled, not run on hardware
//
#include <hip/hip_runtime.h>
#include <hip/hip_bf16.h>
#include <math.h>

// ---------------------------------------------------------------------------
// Mamba2 encoder layer for MI455X (gfx1250, wave32, WMMA).
// GEMMs: bf16 x bf16 -> f32 via v_wmma_f32_16x16x32_bf16, 2x2 tiles per wave,
// software-pipelined global->LDS staging.
// ---------------------------------------------------------------------------

typedef __attribute__((ext_vector_type(16))) __bf16 v16bf;
typedef __attribute__((ext_vector_type(8)))  __bf16 v8bf;
typedef __attribute__((ext_vector_type(8)))  float  v8f;

#define D_MODEL   1024
#define D_STATE   64
#define D_CONV    4
#define HEADDIM   64
#define D_INNER   2048
#define NHEADS    32
#define CONV_DIM  2176          // D_INNER + 2*D_STATE
#define D_IN_PROJ 4256          // 2*D_INNER + 2*D_STATE + NHEADS
#define FFN_DIM   4096
#define B_SZ      2
#define SEQ       2048
#define MROWS     (B_SZ * SEQ)  // 4096 token rows

__device__ __forceinline__ float sigmoidf_(float x) { return 1.0f / (1.0f + expf(-x)); }
__device__ __forceinline__ float siluf_(float x)    { return x * sigmoidf_(x); }

// ----------------------------- block reduction -----------------------------
__device__ __forceinline__ float block_reduce_sum_256(float v, float* sbuf) {
    const int t = threadIdx.x;
    sbuf[t] = v;
    __syncthreads();
    for (int s = 128; s > 0; s >>= 1) {
        if (t < s) sbuf[t] += sbuf[t + s];
        __syncthreads();
    }
    float r = sbuf[0];
    __syncthreads();
    return r;
}

// ----------------------------- f32 -> bf16 convert -------------------------
__global__ __launch_bounds__(256) void cvt_f32_bf16_kernel(
    const float* __restrict__ in, __bf16* __restrict__ out, size_t n) {
    size_t i = (size_t)blockIdx.x * 256 + threadIdx.x;
    size_t stride = (size_t)gridDim.x * 256;
    for (; i < n; i += stride) out[i] = (__bf16)in[i];
}

// ----------------------------- LayerNorm -----------------------------------
template <typename OutT>
__global__ __launch_bounds__(256) void layernorm_kernel(
    const float* __restrict__ x, const float* __restrict__ w,
    const float* __restrict__ b, OutT* __restrict__ out, int D) {
    __shared__ float sbuf[256];
    const size_t row = blockIdx.x;
    const float* xr = x + row * (size_t)D;
    float s = 0.0f;
    for (int i = threadIdx.x; i < D; i += 256) s += xr[i];
    const float mean = block_reduce_sum_256(s, sbuf) / (float)D;
    float v = 0.0f;
    for (int i = threadIdx.x; i < D; i += 256) {
        float d = xr[i] - mean;
        v += d * d;
    }
    const float var = block_reduce_sum_256(v, sbuf) / (float)D;
    const float inv = rsqrtf(var + 1e-5f);
    for (int i = threadIdx.x; i < D; i += 256)
        out[row * (size_t)D + i] = (OutT)((xr[i] - mean) * inv * w[i] + b[i]);
}

// ----------------------------- WMMA GEMM -----------------------------------
// C[M,N] = A[M,K] @ B[N,K]^T  (bf16 row-major operands, K contiguous).
// Block tile 128x64, 8 waves as 4(M) x 2(N); each wave owns a 32x32 tile
// (2x2 WMMA tiles -> 4 v_wmma per K-step). Staging is software-pipelined.
// OP bit0: +bias[col]; bit1: exact GELU; bit2: +res[row*N+col].
#define BM 128
#define BN 64
#define BK 32

template <int OP>
__global__ __launch_bounds__(256) void gemm_wmma_bf16(
    const __bf16* __restrict__ A, const __bf16* __restrict__ Bw,
    const float* __restrict__ bias, const float* __restrict__ res,
    float* __restrict__ C, int M, int N, int K) {
    __shared__ alignas(32) __bf16 As[BM * BK];   // 8 KB
    __shared__ alignas(32) __bf16 Bs[BN * BK];   // 4 KB

    const int t    = threadIdx.x;
    const int lane = t & 31;
    const int w    = t >> 5;            // 8 waves: 4 (M) x 2 (N)
    const int wm   = (w & 3) << 5;      // 0,32,64,96
    const int wn   = (w >> 2) << 5;     // 0,32
    const size_t m0 = (size_t)blockIdx.y * BM;
    const size_t n0 = (size_t)blockIdx.x * BN;

    // staging: 16B (8 x bf16) chunks. A: 512 chunks -> 2/thread; B: 256 -> 1/thread
    const int arow0 = t >> 2;           // 0..63
    const int arow1 = arow0 + 64;       // 64..127
    const int ach   = (t & 3) << 3;     // 0,8,16,24
    const int brow  = t >> 2;           // 0..63
    size_t browg = n0 + brow;
    if (browg >= (size_t)N) browg = (size_t)N - 1;   // clamp (results masked)

    // fragment addressing per CDNA5 VGPR layouts
    const int frm = wm + (lane & 15);
    const int ka  = (lane & 16) ? 8 : 0;   // A: K halves {ka..ka+7, ka+16..ka+23}
    const int frn = wn + (lane & 15);
    const int kb  = (lane & 16) ? 8 : 0;   // B: K halves {kb..kb+7, kb+16..kb+23}

    v8f acc[2][2] = {};

    // -------- software pipeline: preload tile kt=0 into registers --------
    v8bf ra0 = *(const v8bf*)&A[(m0 + arow0) * (size_t)K + ach];
    v8bf ra1 = *(const v8bf*)&A[(m0 + arow1) * (size_t)K + ach];
    v8bf rb  = *(const v8bf*)&Bw[browg * (size_t)K + ach];

    for (int kt = 0; kt < K; kt += BK) {
        *(v8bf*)&As[arow0 * BK + ach] = ra0;
        *(v8bf*)&As[arow1 * BK + ach] = ra1;
        *(v8bf*)&Bs[brow  * BK + ach] = rb;
        __syncthreads();

        const int kn = kt + BK;
        if (kn < K) {  // preload next tile; wait lands after the WMMAs below
            ra0 = *(const v8bf*)&A[(m0 + arow0) * (size_t)K + kn + ach];
            ra1 = *(const v8bf*)&A[(m0 + arow1) * (size_t)K + kn + ach];
            rb  = *(const v8bf*)&Bw[browg * (size_t)K + kn + ach];
            if (kn + BK < K)
                __builtin_prefetch(&A[(m0 + arow0) * (size_t)K + kn + BK + ach], 0, 1);
        }

        // A fragments (rows frm, frm+16)
        v8bf a0lo = *(const v8bf*)&As[frm * BK + ka];
        v8bf a0hi = *(const v8bf*)&As[frm * BK + ka + 16];
        v8bf a1lo = *(const v8bf*)&As[(frm + 16) * BK + ka];
        v8bf a1hi = *(const v8bf*)&As[(frm + 16) * BK + ka + 16];
        v16bf af0 = __builtin_shufflevector(a0lo, a0hi, 0, 1, 2, 3, 4, 5, 6, 7,
                                            8, 9, 10, 11, 12, 13, 14, 15);
        v16bf af1 = __builtin_shufflevector(a1lo, a1hi, 0, 1, 2, 3, 4, 5, 6, 7,
                                            8, 9, 10, 11, 12, 13, 14, 15);
        // B fragments (cols frn, frn+16)
        v16bf bf0 = __builtin_shufflevector(*(const v8bf*)&Bs[frn * BK + kb],
                                            *(const v8bf*)&Bs[frn * BK + kb + 16],
                                            0, 1, 2, 3, 4, 5, 6, 7, 8, 9, 10, 11, 12, 13, 14, 15);
        v16bf bf1 = __builtin_shufflevector(*(const v8bf*)&Bs[(frn + 16) * BK + kb],
                                            *(const v8bf*)&Bs[(frn + 16) * BK + kb + 16],
                                            0, 1, 2, 3, 4, 5, 6, 7, 8, 9, 10, 11, 12, 13, 14, 15);

        acc[0][0] = __builtin_amdgcn_wmma_f32_16x16x32_bf16(
            false, af0, false, bf0, (short)0, acc[0][0], false, false);
        acc[0][1] = __builtin_amdgcn_wmma_f32_16x16x32_bf16(
            false, af0, false, bf1, (short)0, acc[0][1], false, false);
        acc[1][0] = __builtin_amdgcn_wmma_f32_16x16x32_bf16(
            false, af1, false, bf0, (short)0, acc[1][0], false, false);
        acc[1][1] = __builtin_amdgcn_wmma_f32_16x16x32_bf16(
            false, af1, false, bf1, (short)0, acc[1][1], false, false);
        __syncthreads();
    }

    // -------- epilogue (D layout: VGPR v -> M=v for lanes<16, M=v+8 else) ----
    const int mo = (lane & 16) ? 8 : 0;
#pragma unroll
    for (int a = 0; a < 2; ++a) {
        const size_t rowb = m0 + wm + a * 16 + mo;
#pragma unroll
        for (int bb = 0; bb < 2; ++bb) {
            const size_t col = n0 + wn + bb * 16 + (lane & 15);
            if (col < (size_t)N) {
                const float bv = (OP & 1) ? bias[col] : 0.0f;
#pragma unroll
                for (int v = 0; v < 8; ++v) {
                    float val = acc[a][bb][v] + bv;
                    if (OP & 2)
                        val = 0.5f * val * (1.0f + erff(val * 0.70710678118654752f));
                    if (OP & 4) val += res[(rowb + v) * (size_t)N + col];
                    C[(rowb + v) * (size_t)N + col] = val;
                }
            }
        }
    }
}

// ----------------------------- dt = softplus(dt + bias) --------------------
__global__ __launch_bounds__(256) void dt_softplus_kernel(
    const float* __restrict__ zx, const float* __restrict__ dt_bias,
    float* __restrict__ dt_out) {
    const size_t idx = (size_t)blockIdx.x * 256 + threadIdx.x;  // MROWS*NHEADS
    if (idx >= (size_t)MROWS * NHEADS) return;
    const int h = (int)(idx & (NHEADS - 1));
    const size_t row = idx >> 5;
    float x = zx[row * D_IN_PROJ + (2 * D_INNER + 2 * D_STATE) + h] + dt_bias[h];
    dt_out[idx] = (x > 20.0f) ? x : log1pf(expf(x));
}

// ----------------------------- causal dwconv + SiLU ------------------------
__global__ __launch_bounds__(256) void conv_silu_kernel(
    const float* __restrict__ zx, const float* __restrict__ cw,
    const float* __restrict__ cb, float* __restrict__ out) {
    const size_t idx = (size_t)blockIdx.x * 256 + threadIdx.x;  // MROWS*CONV_DIM
    if (idx >= (size_t)MROWS * CONV_DIM) return;
    const int c = (int)(idx % CONV_DIM);
    const size_t bt = idx / CONV_DIM;
    const int tpos = (int)(bt % SEQ);
    const size_t bb = bt / SEQ;
    float acc = cb[c];
#pragma unroll
    for (int j = 0; j < D_CONV; ++j) {
        const int tt = tpos - (D_CONV - 1) + j;
        if (tt >= 0)
            acc += cw[c * D_CONV + j] *
                   zx[(bb * SEQ + tt) * (size_t)D_IN_PROJ + D_INNER + c];
    }
    out[idx] = siluf_(acc);
}

// ----------------------------- SSM sequential scan -------------------------
// One block per (batch, head): 64 lanes each own one HEADDIM row p of the
// 64x64 state, kept in 64 VGPRs; B/C broadcast through LDS each step.
__global__ __launch_bounds__(64) void ssm_scan_kernel(
    const float* __restrict__ xbc, const float* __restrict__ dtb,
    const float* __restrict__ A_log, const float* __restrict__ D_param,
    float* __restrict__ y) {
    const int h = blockIdx.x & (NHEADS - 1);
    const int b = blockIdx.x >> 5;
    const int p = threadIdx.x;  // 0..63
    __shared__ float Bsh[D_STATE];
    __shared__ float Csh[D_STATE];

    float hs[D_STATE];
#pragma unroll
    for (int n = 0; n < D_STATE; ++n) hs[n] = 0.0f;

    const float Ah = -expf(A_log[h]);
    const float Dh = D_param[h];

    for (int t = 0; t < SEQ; ++t) {
        const size_t r    = (size_t)b * SEQ + t;
        const size_t xoff = r * CONV_DIM;
        Bsh[p] = xbc[xoff + D_INNER + p];
        Csh[p] = xbc[xoff + D_INNER + D_STATE + p];
        __syncthreads();

        const float xv   = xbc[xoff + h * HEADDIM + p];
        const float dtv  = dtb[r * NHEADS + h];
        const float dA   = expf(dtv * Ah);
        const float coef = dtv * xv;
        float accy = 0.0f;
#pragma unroll
        for (int n = 0; n < D_STATE; ++n) {
            hs[n] = hs[n] * dA + coef * Bsh[n];
            accy += hs[n] * Csh[n];
        }
        y[r * D_INNER + h * HEADDIM + p] = accy + Dh * xv;
        __syncthreads();
    }
}

// ----------------------------- gated RMSNorm (writes bf16) -----------------
__global__ __launch_bounds__(256) void gated_rmsnorm_kernel(
    const float* __restrict__ y, const float* __restrict__ zx,
    const float* __restrict__ nw, __bf16* __restrict__ out) {
    __shared__ float sbuf[256];
    const size_t row = blockIdx.x;
    const float* yr = y + row * (size_t)D_INNER;
    const float* zr = zx + row * (size_t)D_IN_PROJ;  // z = first D_INNER cols
    float v[8];
    float ss = 0.0f;
#pragma unroll
    for (int j = 0; j < 8; ++j) {
        const int i = threadIdx.x + j * 256;
        const float val = yr[i] * siluf_(zr[i]);
        v[j] = val;
        ss += val * val;
    }
    const float tot = block_reduce_sum_256(ss, sbuf);
    const float inv = rsqrtf(tot / (float)D_INNER + 1e-5f);
#pragma unroll
    for (int j = 0; j < 8; ++j) {
        const int i = threadIdx.x + j * 256;
        out[row * (size_t)D_INNER + i] = (__bf16)(v[j] * inv * nw[i]);
    }
}

// ---------------------------------------------------------------------------
extern "C" void kernel_launch(void* const* d_in, const int* in_sizes, int n_in,
                              void* d_out, int out_size, void* d_ws, size_t ws_size,
                              hipStream_t stream) {
    const float* x         = (const float*)d_in[0];
    const float* ln0_w     = (const float*)d_in[1];
    const float* ln0_b     = (const float*)d_in[2];
    const float* ln1_w     = (const float*)d_in[3];
    const float* ln1_b     = (const float*)d_in[4];
    const float* ln2_w     = (const float*)d_in[5];
    const float* ln2_b     = (const float*)d_in[6];
    const float* in_proj_w = (const float*)d_in[7];    // (4256, 1024)
    const float* conv_w    = (const float*)d_in[8];    // (2176, 4)
    const float* conv_b    = (const float*)d_in[9];
    const float* dt_bias   = (const float*)d_in[10];
    const float* A_log     = (const float*)d_in[11];
    const float* D_param   = (const float*)d_in[12];
    const float* norm_w    = (const float*)d_in[13];
    const float* out_proj_w= (const float*)d_in[14];   // (1024, 2048)
    const float* ffn_w1    = (const float*)d_in[15];   // (4096, 1024)
    const float* ffn_b1    = (const float*)d_in[16];
    const float* ffn_w2    = (const float*)d_in[17];   // (1024, 4096)
    const float* ffn_b2    = (const float*)d_in[18];
    float* out = (float*)d_out;

    // --------- workspace carve-up (256B aligned) ----------
    char* ws = (char*)d_ws;
    size_t off = 0;
    auto alloc = [&](size_t bytes) -> char* {
        char* p = ws + off;
        off = (off + bytes + 255) & ~(size_t)255;
        return p;
    };
    float*  xr    = (float*) alloc((size_t)MROWS * D_MODEL   * 4);  // ln0 output / residual
    float*  x2    = (float*) alloc((size_t)MROWS * D_MODEL   * 4);  // after mamba residual
    __bf16* x1b   = (__bf16*)alloc((size_t)MROWS * D_MODEL   * 2);  // ln1 out bf16
    __bf16* hb    = (__bf16*)alloc((size_t)MROWS * D_MODEL   * 2);  // ln2 out bf16
    __bf16* w_in  = (__bf16*)alloc((size_t)D_IN_PROJ * D_MODEL * 2);
    __bf16* w_out = (__bf16*)alloc((size_t)D_MODEL * D_INNER * 2);
    __bf16* w_f1  = (__bf16*)alloc((size_t)FFN_DIM * D_MODEL * 2);
    __bf16* w_f2  = (__bf16*)alloc((size_t)D_MODEL * FFN_DIM * 2);
    float*  zx    = (float*) alloc((size_t)MROWS * D_IN_PROJ * 4);  // reused as h1 (f32)
    float*  dtb   = (float*) alloc((size_t)MROWS * NHEADS    * 4);
    float*  xbc   = (float*) alloc((size_t)MROWS * CONV_DIM  * 4);  // reused as h1b
    float*  ybuf  = (float*) alloc((size_t)MROWS * D_INNER   * 4);
    __bf16* ynb   = (__bf16*)alloc((size_t)MROWS * D_INNER   * 2);
    float*  h1    = zx;                 // (MROWS, FFN_DIM) f32 fits in zx buffer
    __bf16* h1b   = (__bf16*)xbc;       // (MROWS, FFN_DIM) bf16 fits in xbc buffer
    (void)ws_size; (void)n_in; (void)in_sizes; (void)out_size;

    // --------- weight conversions (f32 -> bf16) ----------
    {
        const size_t n1 = (size_t)D_IN_PROJ * D_MODEL;
        const size_t n2 = (size_t)D_MODEL * D_INNER;
        const size_t n3 = (size_t)FFN_DIM * D_MODEL;
        const size_t n4 = (size_t)D_MODEL * FFN_DIM;
        cvt_f32_bf16_kernel<<<(n1 + 255) / 256, 256, 0, stream>>>(in_proj_w,  w_in,  n1);
        cvt_f32_bf16_kernel<<<(n2 + 255) / 256, 256, 0, stream>>>(out_proj_w, w_out, n2);
        cvt_f32_bf16_kernel<<<(n3 + 255) / 256, 256, 0, stream>>>(ffn_w1,     w_f1,  n3);
        cvt_f32_bf16_kernel<<<(n4 + 255) / 256, 256, 0, stream>>>(ffn_w2,     w_f2,  n4);
    }

    // 1) x = layernorm(x, ln0)            -> xr (f32)
    layernorm_kernel<float><<<MROWS, 256, 0, stream>>>(x, ln0_w, ln0_b, xr, D_MODEL);
    // 2) ln1(xr)                          -> x1b (bf16)
    layernorm_kernel<__bf16><<<MROWS, 256, 0, stream>>>(xr, ln1_w, ln1_b, x1b, D_MODEL);
    // 3) zxbcdt = x1 @ in_proj_w.T        (4096 x 4256 x 1024)
    gemm_wmma_bf16<0><<<dim3((D_IN_PROJ + BN - 1) / BN, MROWS / BM), 256, 0, stream>>>(
        x1b, w_in, nullptr, nullptr, zx, MROWS, D_IN_PROJ, D_MODEL);
    // 4) dt = softplus(dt + dt_bias)
    dt_softplus_kernel<<<((size_t)MROWS * NHEADS + 255) / 256, 256, 0, stream>>>(
        zx, dt_bias, dtb);
    // 5) xBC = silu(causal_dwconv(xBC))
    conv_silu_kernel<<<((size_t)MROWS * CONV_DIM + 255) / 256, 256, 0, stream>>>(
        zx, conv_w, conv_b, xbc);
    // 6) sequential SSM scan -> y (+ D*xs fused)
    ssm_scan_kernel<<<B_SZ * NHEADS, 64, 0, stream>>>(xbc, dtb, A_log, D_param, ybuf);
    // 7) y *= silu(z); RMSNorm * norm_w   -> ynb (bf16)
    gated_rmsnorm_kernel<<<MROWS, 256, 0, stream>>>(ybuf, zx, norm_w, ynb);
    // 8) x2 = xr + ynb @ out_proj_w.T     (residual fused into GEMM epilogue)
    gemm_wmma_bf16<4><<<dim3(D_MODEL / BN, MROWS / BM), 256, 0, stream>>>(
        ynb, w_out, nullptr, xr, x2, MROWS, D_MODEL, D_INNER);
    // 9) h = layernorm(x2, ln2)           -> hb (bf16)
    layernorm_kernel<__bf16><<<MROWS, 256, 0, stream>>>(x2, ln2_w, ln2_b, hb, D_MODEL);
    // 10) h1 = gelu(h @ ffn_w1.T + b1)    (bias+GELU fused)
    gemm_wmma_bf16<3><<<dim3(FFN_DIM / BN, MROWS / BM), 256, 0, stream>>>(
        hb, w_f1, ffn_b1, nullptr, h1, MROWS, FFN_DIM, D_MODEL);
    // 11) convert h1 -> bf16
    cvt_f32_bf16_kernel<<<((size_t)MROWS * FFN_DIM + 255) / 256, 256, 0, stream>>>(
        h1, h1b, (size_t)MROWS * FFN_DIM);
    // 12) out = x2 + h1b @ ffn_w2.T + b2  (bias + residual fused)
    gemm_wmma_bf16<5><<<dim3(D_MODEL / BN, MROWS / BM), 256, 0, stream>>>(
        h1b, w_f2, ffn_b2, x2, out, MROWS, D_MODEL, FFN_DIM);
}
